// dgcnnFeat_68075231641914
// MI455X (gfx1250) — matrix operand order repaired and checked by reading, not compiled, not run on hardware
//
#include <hip/hip_runtime.h>
#include <hip/hip_bf16.h>
#include <stdint.h>

typedef __attribute__((ext_vector_type(16))) __bf16 v16bf;
typedef __attribute__((ext_vector_type(8)))  float  v8f;

#define BEV 32
#define PPE 2048
#define KNN 20

union FragU { v16bf v; uint4 q[2]; };

__device__ __forceinline__ v16bf ld_frag2(const uint4* p0, const uint4* p1) {
  FragU f; f.q[0] = *p0; f.q[1] = *p1; return f.v;
}

__device__ __forceinline__ unsigned lds_addr32(const void* p) {
  return (unsigned)(unsigned long long)p;   // generic->LDS: low 32 bits are LDS byte offset
}

// ---------------------------------------------------------------------------
// Weight swizzle: B-fragment layout for v_wmma_f32_16x16x32_bf16.
// dst[((s*NS+n)*32+lane)*16 + e2] = W[k][col], k = 32*s + 16*(lane>>4) + e2,
// col = 16*n + (lane&15). Lets each lane fetch a fragment as 2x ds_load_b128.
// ---------------------------------------------------------------------------
__device__ __forceinline__ void swz_weights(__bf16* dst, const float* W,
                                            int Cin, int Cout, int KS, int NS,
                                            int tid, int nthr) {
  int total = KS * NS * 32 * 16;
  for (int i = tid; i < total; i += nthr) {
    int e2   = i & 15;
    int lane = (i >> 4) & 31;
    int sn   = i >> 9;
    int n    = sn % NS;
    int s    = sn / NS;
    int kh = lane >> 4, nl = lane & 15;
    int k   = 32 * s + 16 * kh + e2;
    int col = 16 * n + nl;
    float v = (k < Cin) ? W[k * Cout + col] : 0.0f;
    dst[i] = (__bf16)v;
  }
}

// Fused BN (eval): out = relu(acc + lb) * ls + lb2
__device__ __forceinline__ void bn_prep(float* lb, float* ls, float* lb2,
                                        const float* b, const float* g,
                                        const float* be, const float* m,
                                        const float* v, int C, int tid, int nthr) {
  for (int i = tid; i < C; i += nthr) {
    float s = g[i] * rsqrtf(v[i] + 1e-5f);
    lb[i]  = b[i];
    ls[i]  = s;
    lb2[i] = be[i] - m[i] * s;
  }
}

// Pack fp32 x ([N][7]) into async-friendly bf16 rows ([N][8], col7 = 0).
__global__ void to_bf8_kernel(const float* __restrict__ x, __bf16* __restrict__ xbf) {
  int i = blockIdx.x * 256 + threadIdx.x;
  if (i < BEV * PPE * 8) {
    int r = i >> 3, c = i & 7;
    xbf[i] = (c < 7) ? (__bf16)x[r * 7 + c] : (__bf16)0.0f;
  }
}

// ---------------------------------------------------------------------------
// kNN kernel (async, double-buffered): 128 queries/block of one event.
// Candidate tiles stream global->LDS via global_load_async_to_lds_b128 while
// the previous tile is WMMA'd (bf16 Gram) and top-20-scanned.
//   C    = true feature count, SRCW = bf16 source row width (16B multiple),
//   CPAD = LDS row width (multiple of 32 for bf16 WMMA K).
// ---------------------------------------------------------------------------
template<int C, int SRCW, int CPAD>
__global__ __launch_bounds__(128)
void knn_async_kernel(const __bf16* __restrict__ xin, int* __restrict__ idxout) {
  constexpr int KS = CPAD / 32;
  constexpr int CHUNKS = SRCW / 8;      // 16B chunks per source row
  constexpr int NT = PPE / 128;
  extern __shared__ char smem[];
  __bf16* qf  = (__bf16*)smem;              // [128][CPAD]
  __bf16* cf0 = qf + 128 * CPAD;            // [128][CPAD] double-buffered
  __bf16* cf1 = cf0 + 128 * CPAD;
  float*  qsq = (float*)(cf1 + 128 * CPAD); // [128]
  float*  csq = qsq + 128;                  // [128]
  float*  gram = csq + 128;                 // [128][128]

  const int e   = blockIdx.y;
  const int q0  = blockIdx.x * 128;
  const int tid = threadIdx.x;
  const int lane = tid & 31, wid = tid >> 5;
  const int nl = lane & 15, kh = lane >> 4;
  const __bf16* xev = xin + (size_t)e * PPE * SRCW;
  const unsigned long long sbase = (unsigned long long)xev;
  const unsigned qf_b  = lds_addr32(qf);
  const unsigned cf_b0 = lds_addr32(cf0);
  const unsigned cf_b1 = lds_addr32(cf1);

  // Issue async copies: query rows + candidate tile 0.
  for (int i = tid; i < 128 * CHUNKS; i += 128) {
    int row = i / CHUNKS, seg = i % CHUNKS;
    unsigned lds  = qf_b + (unsigned)((row * CPAD + seg * 8) * 2);
    unsigned voff = (unsigned)(((q0 + row) * SRCW + seg * 8) * 2);
    __asm__ volatile("global_load_async_to_lds_b128 %0, %1, %2"
                     :: "v"(lds), "v"(voff), "s"(sbase) : "memory");
  }
  for (int i = tid; i < 128 * CHUNKS; i += 128) {
    int row = i / CHUNKS, seg = i % CHUNKS;
    unsigned lds  = cf_b0 + (unsigned)((row * CPAD + seg * 8) * 2);
    unsigned voff = (unsigned)((row * SRCW + seg * 8) * 2);
    __asm__ volatile("global_load_async_to_lds_b128 %0, %1, %2"
                     :: "v"(lds), "v"(voff), "s"(sbase) : "memory");
  }
  // Zero the K-pad columns once; async copies never touch them.
  if constexpr (CPAD > SRCW) {
    int r = tid;  // 128 threads, 128 rows
    for (int c = SRCW; c < CPAD; ++c) {
      qf[r * CPAD + c]  = (__bf16)0.0f;
      cf0[r * CPAD + c] = (__bf16)0.0f;
      cf1[r * CPAD + c] = (__bf16)0.0f;
    }
  }
  __asm__ volatile("s_wait_asynccnt 0x0" ::: "memory");
  __syncthreads();

  {
    float s = 0.0f;
    #pragma unroll
    for (int c = 0; c < C; ++c) { float v = (float)qf[tid * CPAD + c]; s += v * v; }
    qsq[tid] = s;
  }

  // Persistent A fragments: this wave owns M-subtiles {2*wid, 2*wid+1}.
  v16bf afr[2][KS];
  #pragma unroll
  for (int mi = 0; mi < 2; ++mi) {
    int row = (2 * wid + mi) * 16 + nl;
    const uint4* rp = (const uint4*)(qf + row * CPAD);
    #pragma unroll
    for (int s = 0; s < KS; ++s)
      afr[mi][s] = ld_frag2(rp + 4 * s + kh, rp + 4 * s + kh + 2);
  }

  float dk[KNN]; int ik[KNN];
  #pragma unroll
  for (int j = 0; j < KNN; ++j) { dk[j] = 3.0e38f; ik[j] = 0; }

  for (int ct = 0; ct < NT; ++ct) {
    const __bf16* cf = (ct & 1) ? cf1 : cf0;
    if (ct) {
      __asm__ volatile("s_wait_asynccnt 0x0" ::: "memory");
      __syncthreads();
    }
    // Prefetch next tile into the other buffer (overlaps with compute below).
    if (ct + 1 < NT) {
      unsigned nb = (ct & 1) ? cf_b0 : cf_b1;
      for (int i = tid; i < 128 * CHUNKS; i += 128) {
        int row = i / CHUNKS, seg = i % CHUNKS;
        unsigned lds  = nb + (unsigned)((row * CPAD + seg * 8) * 2);
        unsigned voff = (unsigned)((((ct + 1) * 128 + row) * SRCW + seg * 8) * 2);
        __asm__ volatile("global_load_async_to_lds_b128 %0, %1, %2"
                         :: "v"(lds), "v"(voff), "s"(sbase) : "memory");
      }
    }
    {
      float s = 0.0f;
      #pragma unroll
      for (int c = 0; c < C; ++c) { float v = (float)cf[tid * CPAD + c]; s += v * v; }
      csq[tid] = s;
    }

    // 128x128 Gram tile: Q (A) x Cand^T (B)
    for (int n = 0; n < 8; ++n) {
      v16bf bfr[KS];
      const uint4* bp = (const uint4*)(cf + (n * 16 + nl) * CPAD);
      #pragma unroll
      for (int s = 0; s < KS; ++s)
        bfr[s] = ld_frag2(bp + 4 * s + kh, bp + 4 * s + kh + 2);
      #pragma unroll
      for (int mi = 0; mi < 2; ++mi) {
        v8f acc = {};
        #pragma unroll
        for (int s = 0; s < KS; ++s)
          acc = __builtin_amdgcn_wmma_f32_16x16x32_bf16(
              false, afr[mi][s], false, bfr[s], (short)0, acc, false, false);
        int mrow = (2 * wid + mi) * 16 + 8 * kh;
        #pragma unroll
        for (int r = 0; r < 8; ++r)
          gram[(mrow + r) * 128 + n * 16 + nl] = acc[r];
      }
    }
    __syncthreads();

    // Rotated scan avoids 32-way LDS bank conflicts on gram columns.
    {
      float qs = qsq[tid];
      for (int jj = 0; jj < 128; ++jj) {
        int j = (jj + tid) & 127;
        float d = qs + csq[j] - 2.0f * gram[tid * 128 + j];
        if (d < dk[KNN - 1]) {
          dk[KNN - 1] = d; ik[KNN - 1] = ct * 128 + j;
          #pragma unroll
          for (int s = KNN - 1; s > 0; --s) {
            if (dk[s] < dk[s - 1]) {
              float td = dk[s]; dk[s] = dk[s - 1]; dk[s - 1] = td;
              int   ti = ik[s]; ik[s] = ik[s - 1]; ik[s - 1] = ti;
            }
          }
        }
      }
    }
    // No trailing barrier: next iteration's wait+barrier orders buffer reuse.
  }

  int* op = idxout + (size_t)(e * PPE + q0 + tid) * KNN;
  #pragma unroll
  for (int j = 0; j < KNN; ++j) op[j] = ik[j];
}

// ---------------------------------------------------------------------------
// One MLP layer over a wave-private 80-row tile (5 M-subtiles), bf16 WMMA.
// ---------------------------------------------------------------------------
template<int KS, int NS, int STRIDE>
__device__ __forceinline__ void mlp_layer(const __bf16* src, __bf16* dst,
                                          const __bf16* wswz,
                                          const float* lb, const float* ls,
                                          const float* lb2, int lane) {
  const int nl = lane & 15, kh = lane >> 4;
  for (int t = 0; t < 5; ++t) {
    v16bf a[KS];
    const uint4* rp = (const uint4*)(src + (t * 16 + nl) * STRIDE);
    #pragma unroll
    for (int s = 0; s < KS; ++s)
      a[s] = ld_frag2(rp + 4 * s + 2 * kh, rp + 4 * s + 2 * kh + 1);
    #pragma unroll
    for (int n = 0; n < NS; ++n) {
      v8f acc = {};
      #pragma unroll
      for (int s = 0; s < KS; ++s) {
        const uint4* wp = (const uint4*)(wswz + (((s * NS + n) * 32 + lane) << 4));
        v16bf b = ld_frag2(wp, wp + 1);
        acc = __builtin_amdgcn_wmma_f32_16x16x32_bf16(
            false, a[s], false, b, (short)0, acc, false, false);
      }
      int col = n * 16 + nl;
      float sc = ls[col], bb = lb[col], b2 = lb2[col];
      #pragma unroll
      for (int r = 0; r < 8; ++r) {
        int m = t * 16 + r + 8 * kh;
        float v = acc[r] + bb;
        v = fmaxf(v, 0.0f);
        v = v * sc + b2;
        dst[m * STRIDE + col] = (__bf16)v;
      }
    }
  }
}

// Final layer: BN epilogue + max-aggregation over 20 edges via ds_max_num_f32.
template<int KS, int NS, int STRIDE, int COUT>
__device__ __forceinline__ void mlp_layer_max(const __bf16* src, float* pmax,
                                              const __bf16* wswz,
                                              const float* lb, const float* ls,
                                              const float* lb2, int lane) {
  const int nl = lane & 15, kh = lane >> 4;
  for (int t = 0; t < 5; ++t) {
    v16bf a[KS];
    const uint4* rp = (const uint4*)(src + (t * 16 + nl) * STRIDE);
    #pragma unroll
    for (int s = 0; s < KS; ++s)
      a[s] = ld_frag2(rp + 4 * s + 2 * kh, rp + 4 * s + 2 * kh + 1);
    #pragma unroll
    for (int n = 0; n < NS; ++n) {
      v8f acc = {};
      #pragma unroll
      for (int s = 0; s < KS; ++s) {
        const uint4* wp = (const uint4*)(wswz + (((s * NS + n) * 32 + lane) << 4));
        v16bf b = ld_frag2(wp, wp + 1);
        acc = __builtin_amdgcn_wmma_f32_16x16x32_bf16(
            false, a[s], false, b, (short)0, acc, false, false);
      }
      int col = n * 16 + nl;
      float sc = ls[col], bb = lb[col], b2 = lb2[col];
      #pragma unroll
      for (int r = 0; r < 8; ++r) {
        int m = t * 16 + r + 8 * kh;
        float v = acc[r] + bb;
        v = fmaxf(v, 0.0f);
        v = v * sc + b2;
        float* pp = pmax + (m / 20) * COUT + col;
        __asm__ volatile("ds_max_num_f32 %0, %1"
                         :: "v"(lds_addr32(pp)), "v"(v) : "memory");
      }
    }
  }
}

// ---------------------------------------------------------------------------
// EdgeConv1: gather [x_i, x_j-x_i] (14d) -> MLP 14->64->64->64 -> max over 20.
// Block = 4 waves * 4 points each = 16 points.
// ---------------------------------------------------------------------------
__global__ __launch_bounds__(128)
void edgeconv1_kernel(const float* __restrict__ x, const int* __restrict__ knn,
                      const float* __restrict__ W1, const float* __restrict__ B1,
                      const float* __restrict__ G1, const float* __restrict__ E1,
                      const float* __restrict__ M1, const float* __restrict__ V1,
                      const float* __restrict__ W2, const float* __restrict__ B2,
                      const float* __restrict__ G2, const float* __restrict__ E2,
                      const float* __restrict__ M2, const float* __restrict__ V2,
                      const float* __restrict__ W3, const float* __restrict__ B3,
                      const float* __restrict__ G3, const float* __restrict__ E3,
                      const float* __restrict__ M3, const float* __restrict__ V3,
                      __bf16* __restrict__ x1bf, float* __restrict__ dout) {
  extern __shared__ char smem[];
  __bf16* w1s = (__bf16*)smem;           // 1*4*32*16 = 2048
  __bf16* w2s = w1s + 2048;              // 2*4*32*16 = 4096
  __bf16* w3s = w2s + 4096;              // 4096
  float*  bn  = (float*)(w3s + 4096);    // 3 layers * 3 arrays * 64
  __bf16* bufs = (__bf16*)(bn + 9 * 64); // per wave: 2 * 80*64 bf16
  float*  pmaxAll = (float*)(bufs + 4 * 2 * 80 * 64); // per wave: 4*64 f32

  const int tid = threadIdx.x;
  const int lane = tid & 31, wid = tid >> 5;
  const int e = blockIdx.y;
  const int pbase = blockIdx.x * 16;

  swz_weights(w1s, W1, 14, 64, 1, 4, tid, 128);
  swz_weights(w2s, W2, 64, 64, 2, 4, tid, 128);
  swz_weights(w3s, W3, 64, 64, 2, 4, tid, 128);
  bn_prep(bn + 0,   bn + 64,  bn + 128, B1, G1, E1, M1, V1, 64, tid, 128);
  bn_prep(bn + 192, bn + 256, bn + 320, B2, G2, E2, M2, V2, 64, tid, 128);
  bn_prep(bn + 384, bn + 448, bn + 512, B3, G3, E3, M3, V3, 64, tid, 128);
  __syncthreads();

  __bf16* bufA = bufs + wid * 2 * 80 * 64;
  __bf16* bufB = bufA + 80 * 64;
  float*  pmax = pmaxAll + wid * 4 * 64;

  // Gather edge features for this wave's 4 points (80 edges), fp32 source.
  const int* myidx = knn + (size_t)(e * PPE + pbase + wid * 4) * KNN;
  for (int r = lane; r < 80; r += 32) {
    int pi = r / 20;
    int p = pbase + wid * 4 + pi;
    int nbr = myidx[r];
    const float* xi = x + (size_t)(e * PPE + p) * 7;
    const float* xj = x + (size_t)(e * PPE + nbr) * 7;
    __bf16* row = bufA + r * 64;
    #pragma unroll
    for (int c = 0; c < 7; ++c) {
      float a = xi[c], b = xj[c];
      row[c]     = (__bf16)a;
      row[7 + c] = (__bf16)(b - a);
    }
    #pragma unroll
    for (int c = 14; c < 32; ++c) row[c] = (__bf16)0.0f;
  }
  for (int i = lane; i < 4 * 64; i += 32) pmax[i] = -3.0e38f;

  mlp_layer<1, 4, 64>(bufA, bufB, w1s, bn + 0,   bn + 64,  bn + 128, lane);
  mlp_layer<2, 4, 64>(bufB, bufA, w2s, bn + 192, bn + 256, bn + 320, lane);
  mlp_layer_max<2, 4, 64, 64>(bufA, pmax, w3s, bn + 384, bn + 448, bn + 512, lane);

  __asm__ volatile("s_wait_dscnt 0x0" ::: "memory");
  for (int i = lane; i < 4 * 64; i += 32) {
    int pi = i >> 6, col = i & 63;
    float v = pmax[pi * 64 + col];
    size_t g = (size_t)(e * PPE + pbase + wid * 4 + pi);
    x1bf[g * 64 + col] = (__bf16)v;   // bf16 mirror for kNN2 / EdgeConv2
    dout[g * 192 + col] = v;          // fp32 output
  }
}

// ---------------------------------------------------------------------------
// EdgeConv2: gather [x1_i, x1_j-x1_i] (128d, bf16 source) -> 128->128 -> max.
// ---------------------------------------------------------------------------
__global__ __launch_bounds__(128)
void edgeconv2_kernel(const __bf16* __restrict__ x1bf, const int* __restrict__ knn,
                      const float* __restrict__ W, const float* __restrict__ B,
                      const float* __restrict__ G, const float* __restrict__ E,
                      const float* __restrict__ M, const float* __restrict__ V,
                      float* __restrict__ dout) {
  extern __shared__ char smem[];
  __bf16* ws  = (__bf16*)smem;             // 4*8*32*16 = 16384
  float*  bn  = (float*)(ws + 16384);      // 3 * 128
  __bf16* bufs = (__bf16*)(bn + 3 * 128);  // per wave: 80*128 bf16
  float*  pmaxAll = (float*)(bufs + 4 * 80 * 128); // per wave: 4*128 f32

  const int tid = threadIdx.x;
  const int lane = tid & 31, wid = tid >> 5;
  const int e = blockIdx.y;
  const int pbase = blockIdx.x * 16;

  swz_weights(ws, W, 128, 128, 4, 8, tid, 128);
  bn_prep(bn, bn + 128, bn + 256, B, G, E, M, V, 128, tid, 128);
  __syncthreads();

  __bf16* bufA = bufs + wid * 80 * 128;
  float*  pmax = pmaxAll + wid * 4 * 128;

  const int* myidx = knn + (size_t)(e * PPE + pbase + wid * 4) * KNN;
  for (int r = lane; r < 80; r += 32) {
    int pi = r / 20;
    int p = pbase + wid * 4 + pi;
    int nbr = myidx[r];
    const __bf16* xi = x1bf + (size_t)(e * PPE + p) * 64;
    const __bf16* xj = x1bf + (size_t)(e * PPE + nbr) * 64;
    __bf16* row = bufA + r * 128;
    #pragma unroll 8
    for (int c = 0; c < 64; ++c) {
      float a = (float)xi[c], b = (float)xj[c];
      row[c]      = xi[c];
      row[64 + c] = (__bf16)(b - a);
    }
  }
  for (int i = lane; i < 4 * 128; i += 32) pmax[i] = -3.0e38f;

  mlp_layer_max<4, 8, 128, 128>(bufA, pmax, ws, bn, bn + 128, bn + 256, lane);

  __asm__ volatile("s_wait_dscnt 0x0" ::: "memory");
  for (int i = lane; i < 4 * 128; i += 32) {
    int pi = i >> 7, col = i & 127;
    float v = pmax[pi * 128 + col];
    size_t g = (size_t)(e * PPE + pbase + wid * 4 + pi);
    dout[g * 192 + 64 + col] = v;
  }
}

__global__ void copy_batch_kernel(const int* __restrict__ b, int* __restrict__ o, int n) {
  int i = blockIdx.x * 256 + threadIdx.x;
  if (i < n) o[i] = b[i];
}

// ---------------------------------------------------------------------------
extern "C" void kernel_launch(void* const* d_in, const int* in_sizes, int n_in,
                              void* d_out, int out_size, void* d_ws, size_t ws_size,
                              hipStream_t stream) {
  const float* x     = (const float*)d_in[0];
  const int*   batch = (const int*)d_in[1];
  const float *w10 = (const float*)d_in[2],  *b10 = (const float*)d_in[3],
              *g10 = (const float*)d_in[4],  *e10 = (const float*)d_in[5],
              *m10 = (const float*)d_in[6],  *v10 = (const float*)d_in[7];
  const float *w11 = (const float*)d_in[8],  *b11 = (const float*)d_in[9],
              *g11 = (const float*)d_in[10], *e11 = (const float*)d_in[11],
              *m11 = (const float*)d_in[12], *v11 = (const float*)d_in[13];
  const float *w12 = (const float*)d_in[14], *b12 = (const float*)d_in[15],
              *g12 = (const float*)d_in[16], *e12 = (const float*)d_in[17],
              *m12 = (const float*)d_in[18], *v12 = (const float*)d_in[19];
  const float *w20 = (const float*)d_in[20], *b20 = (const float*)d_in[21],
              *g20 = (const float*)d_in[22], *e20 = (const float*)d_in[23],
              *m20 = (const float*)d_in[24], *v20 = (const float*)d_in[25];

  float* out = (float*)d_out;
  const size_t N = (size_t)BEV * PPE;
  __bf16* xbf8 = (__bf16*)d_ws;                         // [N][8]  bf16 (1 MB)
  __bf16* x1bf = (__bf16*)((char*)d_ws + N * 8 * 2);    // [N][64] bf16 (8 MB)
  int*    knn  = (int*)((char*)d_ws + N * 8 * 2 + N * 64 * 2); // [N][20] i32

  const size_t lds_knn1 = 3 * 128 * 32 * sizeof(__bf16) + 2 * 128 * sizeof(float)
                        + 128 * 128 * sizeof(float);    //  91,136 B
  const size_t lds_knn2 = 3 * 128 * 64 * sizeof(__bf16) + 2 * 128 * sizeof(float)
                        + 128 * 128 * sizeof(float);    // 115,712 B
  const size_t lds_m1 = (2048 + 4096 + 4096) * sizeof(__bf16) + 9 * 64 * sizeof(float)
                      + 4 * 2 * 80 * 64 * sizeof(__bf16) + 4 * 4 * 64 * sizeof(float);
  const size_t lds_m2 = 16384 * sizeof(__bf16) + 3 * 128 * sizeof(float)
                      + 4 * 80 * 128 * sizeof(__bf16) + 4 * 4 * 128 * sizeof(float);

  // 0) bf16 pack of raw features (async-copy friendly 16B rows)
  {
    int n = BEV * PPE * 8;
    to_bf8_kernel<<<(n + 255) / 256, 256, 0, stream>>>(x, xbf8);
  }
  // 1) kNN on 7-d features (pad 8 -> K=32)
  knn_async_kernel<7, 8, 32><<<dim3(PPE / 128, BEV), 128, lds_knn1, stream>>>(xbf8, knn);
  // 2) EdgeConv1: writes x1bf (ws) and out[:, 0:64]
  edgeconv1_kernel<<<dim3(PPE / 16, BEV), 128, lds_m1, stream>>>(
      x, knn, w10, b10, g10, e10, m10, v10,
      w11, b11, g11, e11, m11, v11,
      w12, b12, g12, e12, m12, v12, x1bf, out);
  // 3) kNN on 64-d x1
  knn_async_kernel<64, 64, 64><<<dim3(PPE / 128, BEV), 128, lds_knn2, stream>>>(x1bf, knn);
  // 4) EdgeConv2: writes out[:, 64:192]
  edgeconv2_kernel<<<dim3(PPE / 16, BEV), 128, lds_m2, stream>>>(
      x1bf, knn, w20, b20, g20, e20, m20, v20, out);
  // 5) Tuple tail: batch vector
  if (out_size >= BEV * PPE * 192 + BEV * PPE) {
    int n = BEV * PPE;
    copy_batch_kernel<<<(n + 255) / 256, 256, 0, stream>>>(
        batch, (int*)(out + (size_t)BEV * PPE * 192), n);
  }
}
